// PredictiveCodingNet_28733331210547
// MI455X (gfx1250) — matrix operand order repaired and checked by reading, not compiled, not run on hardware
//
#include <hip/hip_runtime.h>

typedef _Float16 h16;
typedef h16  v16h __attribute__((ext_vector_type(16)));
typedef h16  v8h  __attribute__((ext_vector_type(8)));
typedef float v8f __attribute__((ext_vector_type(8)));
typedef float v4f __attribute__((ext_vector_type(4)));

#define IN_DIM 784
#define H1 128
#define H2 64
#define H3 32
#define NOUT 10
#define NTHREADS 256
#define WAVES 8
#define ROWS_PER_BLOCK (WAVES * 16)

// Padded LDS row strides (f16 elements) for conflict-free ds_load_b128:
// bytes/row = 272 / 144 / 80  ->  per-row bank shift 4 / 36 / 20 (mod 64),
// giving 16 disjoint 4-bank spans across the 16 row-lanes of a half-wave.
#define SA1 136   // rows with K-width up to 128
#define SA2 72    // rows with K-width up to 64
#define SA3 40    // rows with K-width up to 32

__device__ __forceinline__ v8f wmma_f16(v16h a, v16h b, v8f c) {
  // v_wmma_f32_16x16x32_f16 : D = A(16x32 f16) x B(32x16 f16) + C(16x16 f32)
  return __builtin_amdgcn_wmma_f32_16x16x32_f16(false, a, false, b, (short)0, c,
                                                false, false);
}

// A fragment (16x32, M = lane&15): per lane 8 f16 at +0 and 8 f16 at +16
__device__ __forceinline__ v16h fragA_lds(const h16* p) {
  v8h lo = *(const v8h*)(p);
  v8h hi = *(const v8h*)(p + 16);
  v16h f;
#pragma unroll
  for (int i = 0; i < 8; ++i) { f[i] = lo[i]; f[8 + i] = hi[i]; }
  return f;
}

// B fragment (32x16, N = lane&15): per lane 8 f16 at +0 and 8 f16 at +16
__device__ __forceinline__ v16h fragB_lds(const h16* p) {
  v8h lo = *(const v8h*)(p);
  v8h hi = *(const v8h*)(p + 16);
  v16h f;
#pragma unroll
  for (int i = 0; i < 8; ++i) { f[i] = lo[i]; f[8 + i] = hi[i]; }
  return f;
}

// A fragment from row-major f32 global (convert in flight); tail => K upper half 0.
// `tail` is wave-uniform, so this branch never splits EXEC.
__device__ __forceinline__ v16h fragA_g32(const float* p, bool tail) {
  v4f a0 = *(const v4f*)(p);
  v4f a1 = *(const v4f*)(p + 4);
  v4f b0 = {0.f, 0.f, 0.f, 0.f}, b1 = {0.f, 0.f, 0.f, 0.f};
  if (!tail) { b0 = *(const v4f*)(p + 16); b1 = *(const v4f*)(p + 20); }
  v16h f;
#pragma unroll
  for (int i = 0; i < 4; ++i) {
    f[i]      = (h16)a0[i];
    f[4 + i]  = (h16)a1[i];
    f[8 + i]  = (h16)b0[i];
    f[12 + i] = (h16)b1[i];
  }
  return f;
}

__device__ __forceinline__ float rmax16(float v) {
#pragma unroll
  for (int m = 1; m < 16; m <<= 1) v = fmaxf(v, __shfl_xor(v, m, 32));
  return v;
}
__device__ __forceinline__ float rsum16(float v) {
#pragma unroll
  for (int m = 1; m < 16; m <<= 1) v += __shfl_xor(v, m, 32);
  return v;
}

__global__ __launch_bounds__(NTHREADS) void pcnet_kernel(
    const float* __restrict__ x,  const float* __restrict__ W1,
    const float* __restrict__ W2, const float* __restrict__ W3,
    const float* __restrict__ W4, const float* __restrict__ L1,
    const float* __restrict__ L2, const float* __restrict__ L3,
    const float* __restrict__ b1, const float* __restrict__ b2,
    const float* __restrict__ b3, const float* __restrict__ b4,
    const int* __restrict__ nstep_p, float* __restrict__ out, int B) {
  // ---- LDS: padded-stride f16 weights + per-wave f16 state staging ----
  __shared__ __align__(16) h16 sL1[H1 * SA1];        // L1^T  (row n, col k)
  __shared__ __align__(16) h16 sW2[H2 * SA1];        // W2 as B^T (row n, col k)
  __shared__ __align__(16) h16 sL2[H2 * SA2];
  __shared__ __align__(16) h16 sW3[H3 * SA2];
  __shared__ __align__(16) h16 sL3[H3 * SA3];
  __shared__ __align__(16) h16 sW4[16 * SA3];        // rows >= NOUT zero-padded
  __shared__ __align__(16) h16 sS1[WAVES * 16 * SA1];  // states; W1 k-slice staging in P phase
  __shared__ __align__(16) h16 sS2[WAVES * 16 * SA2];
  __shared__ __align__(16) h16 sS3[WAVES * 16 * SA3];

  const int tid = threadIdx.x;
  for (int i = tid; i < H2 * H1; i += NTHREADS)
    sW2[(i >> 7) * SA1 + (i & 127)] = (h16)W2[i];
  for (int i = tid; i < H3 * H2; i += NTHREADS)
    sW3[(i >> 6) * SA2 + (i & 63)] = (h16)W3[i];
  for (int i = tid; i < 16 * H3; i += NTHREADS) {
    int rr = i >> 5, cc = i & 31;
    sW4[rr * SA3 + cc] = (rr < NOUT) ? (h16)W4[rr * H3 + cc] : (h16)0.f;
  }
  for (int i = tid; i < H1 * H1; i += NTHREADS) {
    int rr = i >> 7, cc = i & 127;
    sL1[rr * SA1 + cc] = (h16)L1[cc * H1 + rr];   // transpose
  }
  for (int i = tid; i < H2 * H2; i += NTHREADS) {
    int rr = i >> 6, cc = i & 63;
    sL2[rr * SA2 + cc] = (h16)L2[cc * H2 + rr];
  }
  for (int i = tid; i < H3 * H3; i += NTHREADS) {
    int rr = i >> 5, cc = i & 31;
    sL3[rr * SA3 + cc] = (h16)L3[cc * H3 + rr];
  }

  const int lane = tid & 31;
  const int wave = tid >> 5;
  const int r    = lane & 15;        // N (B/C frags) or M (A frags)
  const int kg   = (lane >> 4) & 1;  // half-wave K-group select
  const size_t row0 = (size_t)blockIdx.x * ROWS_PER_BLOCK + (size_t)wave * 16;

  h16* S1w = sS1 + wave * 16 * SA1;
  h16* S2w = sS2 + wave * 16 * SA2;
  h16* S3w = sS3 + wave * 16 * SA3;
  h16* w1s = sS1;                    // 128 x SA3 staging buffer during P phase

  // ---- loop-invariant P = x @ W1^T + b1 (16x128 per wave, in registers) ----
  // Block-cooperatively stage each 128x32 W1 k-slice in LDS (f16), so W1 is
  // fetched from global once per block and the inner body is branch-free.
  v8f P[8];
#pragma unroll
  for (int t = 0; t < 8; ++t) {
    float bv = b1[t * 16 + r];
#pragma unroll
    for (int i = 0; i < 8; ++i) P[t][i] = bv;
  }
  {
    const float* xrow = x + (row0 + (size_t)r) * IN_DIM + kg * 8;
#pragma unroll 1
    for (int kf = 0; kf < 25; ++kf) {            // 784 = 24*32 + 16
      const bool tail = (kf == 24);
      __syncthreads();                           // staging WAR vs previous slice
      if (!tail) {
        for (int e = tid; e < H1 * 32; e += NTHREADS) {
          int rr = e >> 5, cc = e & 31;
          w1s[rr * SA3 + cc] = (h16)W1[(size_t)rr * IN_DIM + kf * 32 + cc];
        }
      } else {
        for (int e = tid; e < H1 * 32; e += NTHREADS) {
          int rr = e >> 5, cc = e & 31;
          w1s[rr * SA3 + cc] =
              (cc < 16) ? (h16)W1[(size_t)rr * IN_DIM + 768 + cc] : (h16)0.f;
        }
      }
      __syncthreads();
      v16h a = fragA_g32(xrow + kf * 32, tail);
#pragma unroll
      for (int t = 0; t < 8; ++t)
        P[t] = wmma_f16(a, fragB_lds(w1s + (t * 16 + r) * SA3 + kg * 8), P[t]);
    }
  }
  __syncthreads();                               // staging done; sS1 becomes state

  // ---- states (f32 accumulators in C-fragment layout) ----
  v8f s1[8], s2[4], s3[2], s4;
#pragma unroll
  for (int t = 0; t < 8; ++t)
#pragma unroll
    for (int i = 0; i < 8; ++i) s1[t][i] = 0.f;
#pragma unroll
  for (int t = 0; t < 4; ++t)
#pragma unroll
    for (int i = 0; i < 8; ++i) s2[t][i] = 0.f;
#pragma unroll
  for (int t = 0; t < 2; ++t)
#pragma unroll
    for (int i = 0; i < 8; ++i) s3[t][i] = 0.f;
#pragma unroll
  for (int i = 0; i < 8; ++i) s4[i] = 0.f;

  float bias2[4], bias3[2];
#pragma unroll
  for (int t = 0; t < 4; ++t) bias2[t] = b2[t * 16 + r];
#pragma unroll
  for (int t = 0; t < 2; ++t) bias3[t] = b3[t * 16 + r];
  const float bias4 = (r < NOUT) ? b4[r] : 0.f;
  const bool  valid = (r < NOUT);

  const int nsteps = nstep_p[0];
#pragma unroll 1
  for (int step = 0; step < nsteps; ++step) {
    // dump old states as f16 into this wave's private LDS region
#pragma unroll
    for (int t = 0; t < 8; ++t) {
      h16* p = S1w + t * 16 + r;
#pragma unroll
      for (int v = 0; v < 8; ++v) p[(v + kg * 8) * SA1] = (h16)s1[t][v];
    }
#pragma unroll
    for (int t = 0; t < 4; ++t) {
      h16* p = S2w + t * 16 + r;
#pragma unroll
      for (int v = 0; v < 8; ++v) p[(v + kg * 8) * SA2] = (h16)s2[t][v];
    }
#pragma unroll
    for (int t = 0; t < 2; ++t) {
      h16* p = S3w + t * 16 + r;
#pragma unroll
      for (int v = 0; v < 8; ++v) p[(v + kg * 8) * SA3] = (h16)s3[t][v];
    }
    // Wave-private region + in-order LDS per wave: only a compiler fence is
    // needed to keep the ds_loads below after the ds_stores above.
    asm volatile("" ::: "memory");

    // old-state A fragments (all cross-layer deps flow through these)
    v16h A1[4], A2[2], A3;
#pragma unroll
    for (int kf = 0; kf < 4; ++kf)
      A1[kf] = fragA_lds(S1w + r * SA1 + kf * 32 + kg * 8);
#pragma unroll
    for (int kf = 0; kf < 2; ++kf)
      A2[kf] = fragA_lds(S2w + r * SA2 + kf * 32 + kg * 8);
    A3 = fragA_lds(S3w + r * SA3 + kg * 8);

    // layer 1: pred1 = P + s1@L1 ; s1 = relu(0.8 s1 + 0.2 pred1)
#pragma unroll
    for (int t = 0; t < 8; ++t) {
      v8f c = P[t];
#pragma unroll
      for (int kf = 0; kf < 4; ++kf)
        c = wmma_f16(A1[kf], fragB_lds(sL1 + (t * 16 + r) * SA1 + kf * 32 + kg * 8), c);
#pragma unroll
      for (int i = 0; i < 8; ++i) s1[t][i] = fmaxf(0.f, 0.8f * s1[t][i] + 0.2f * c[i]);
    }
    // layer 2: pred2 = s1@W2^T + b2 + s2@L2
#pragma unroll
    for (int t = 0; t < 4; ++t) {
      v8f c;
#pragma unroll
      for (int i = 0; i < 8; ++i) c[i] = bias2[t];
#pragma unroll
      for (int kf = 0; kf < 4; ++kf)
        c = wmma_f16(A1[kf], fragB_lds(sW2 + (t * 16 + r) * SA1 + kf * 32 + kg * 8), c);
#pragma unroll
      for (int kf = 0; kf < 2; ++kf)
        c = wmma_f16(A2[kf], fragB_lds(sL2 + (t * 16 + r) * SA2 + kf * 32 + kg * 8), c);
#pragma unroll
      for (int i = 0; i < 8; ++i) s2[t][i] = fmaxf(0.f, 0.8f * s2[t][i] + 0.2f * c[i]);
    }
    // layer 3: pred3 = s2@W3^T + b3 + s3@L3
#pragma unroll
    for (int t = 0; t < 2; ++t) {
      v8f c;
#pragma unroll
      for (int i = 0; i < 8; ++i) c[i] = bias3[t];
#pragma unroll
      for (int kf = 0; kf < 2; ++kf)
        c = wmma_f16(A2[kf], fragB_lds(sW3 + (t * 16 + r) * SA2 + kf * 32 + kg * 8), c);
      c = wmma_f16(A3, fragB_lds(sL3 + (t * 16 + r) * SA3 + kg * 8), c);
#pragma unroll
      for (int i = 0; i < 8; ++i) s3[t][i] = fmaxf(0.f, 0.8f * s3[t][i] + 0.2f * c[i]);
    }
    // layer 4: pred4 = s3@W4^T + b4 ; s4 = softmax(0.8 s4 + 0.2 pred4), cols 0..9
    {
      v8f c;
#pragma unroll
      for (int i = 0; i < 8; ++i) c[i] = bias4;
      c = wmma_f16(A3, fragB_lds(sW4 + r * SA3 + kg * 8), c);
#pragma unroll
      for (int v = 0; v < 8; ++v) {
        float z  = 0.8f * s4[v] + 0.2f * c[v];
        float mx = rmax16(valid ? z : -3.0e38f);
        float e  = valid ? exp2f((z - mx) * 1.44269504f) : 0.f;
        float sm = rsum16(e);
        s4[v] = e / sm;
      }
    }
  }

  // ---- write outputs: concat(s1, s2, s3, s4) flat in f32 ----
  float* o1 = out;
  float* o2 = out + (size_t)B * H1;
  float* o3 = o2 + (size_t)B * H2;
  float* o4 = o3 + (size_t)B * H3;
#pragma unroll
  for (int t = 0; t < 8; ++t)
#pragma unroll
    for (int v = 0; v < 8; ++v)
      o1[(row0 + v + kg * 8) * H1 + t * 16 + r] = s1[t][v];
#pragma unroll
  for (int t = 0; t < 4; ++t)
#pragma unroll
    for (int v = 0; v < 8; ++v)
      o2[(row0 + v + kg * 8) * H2 + t * 16 + r] = s2[t][v];
#pragma unroll
  for (int t = 0; t < 2; ++t)
#pragma unroll
    for (int v = 0; v < 8; ++v)
      o3[(row0 + v + kg * 8) * H3 + t * 16 + r] = s3[t][v];
  if (valid) {
#pragma unroll
    for (int v = 0; v < 8; ++v)
      o4[(row0 + v + kg * 8) * NOUT + r] = s4[v];
  }
}

extern "C" void kernel_launch(void* const* d_in, const int* in_sizes, int n_in,
                              void* d_out, int out_size, void* d_ws, size_t ws_size,
                              hipStream_t stream) {
  const float* x  = (const float*)d_in[0];
  const float* W1 = (const float*)d_in[1];
  const float* W2 = (const float*)d_in[2];
  const float* W3 = (const float*)d_in[3];
  const float* W4 = (const float*)d_in[4];
  const float* L1 = (const float*)d_in[5];
  const float* L2 = (const float*)d_in[6];
  const float* L3 = (const float*)d_in[7];
  const float* b1 = (const float*)d_in[8];
  const float* b2 = (const float*)d_in[9];
  const float* b3 = (const float*)d_in[10];
  const float* b4 = (const float*)d_in[11];
  const int* nst  = (const int*)d_in[12];

  const int B = in_sizes[0] / IN_DIM;          // 65536
  const int grid = B / ROWS_PER_BLOCK;         // 512 blocks of 8 waves x 16 rows

  pcnet_kernel<<<grid, NTHREADS, 0, stream>>>(x, W1, W2, W3, W4, L1, L2, L3,
                                              b1, b2, b3, b4, nst,
                                              (float*)d_out, B);
}
